// WindowAttention1D_50534585205178
// MI455X (gfx1250) — matrix operand order repaired and checked by reading, not compiled, not run on hardware
//
#include <hip/hip_runtime.h>
#include <hip/hip_bf16.h>
#include <math.h>

typedef __attribute__((ext_vector_type(16))) _Float16 v16h;
typedef __attribute__((ext_vector_type(8)))  _Float16 v8h;
typedef __attribute__((ext_vector_type(4)))  _Float16 v4h;
typedef __attribute__((ext_vector_type(8)))  float    v8f;
typedef __attribute__((ext_vector_type(4)))  float    v4f;

#define DIM   256
#define HEADS 8
#define WSZ   64
#define HD    32

// LDS strides (in halfs); all chosen so (row*stride)*2 % 16 == 0
#define XS 264   // x / attn-out staging: 64 x 256 (+8 pad)
#define QS 520   // q|k staging: 64 x 512 (+8 pad)
#define VS 72    // v transposed: 256 x 64 (+8 pad)
#define PS 72    // probs per head: 64 x 64 (+8 pad)

// ---------------------------------------------------------------------------
// Fragment loaders per CDNA5 ISA 7.12.2 (wave32).
// A (16x32 f16): lane l holds row M = l&15; half (l>>4): K = (l>>4)*16 + 0..15
//                (contiguous, matching the B fragments).
__device__ inline v16h load_a_frag(const _Float16* sm, int stride, int mbase,
                                   int kbase, int lane) {
  int row = mbase + (lane & 15);
  int kh  = lane >> 4;
  const _Float16* p0 = sm + row * stride + kbase + kh * 16;
  v8h lo = *(const v8h*)p0;
  v8h hi = *(const v8h*)(p0 + 8);
  v16h r;
#pragma unroll
  for (int i = 0; i < 8; ++i) { r[i] = lo[i]; r[8 + i] = hi[i]; }
  return r;
}

// B (32x16 f16) built from a row-major matrix whose ROWS are the B columns:
// B(k, n) = src[nbase + n][kbase + k].  Lane l: n = l&15, K = (l>>4)*16 + 2j(+1)
// -> 16 contiguous halfs per lane.
__device__ inline v16h load_bt_frag(const _Float16* sm, int stride, int nbase,
                                    int kbase, int lane) {
  int row = nbase + (lane & 15);
  int kh  = lane >> 4;
  const _Float16* p = sm + row * stride + kbase + kh * 16;
  v8h lo = *(const v8h*)p;
  v8h hi = *(const v8h*)(p + 8);
  v16h r;
#pragma unroll
  for (int i = 0; i < 8; ++i) { r[i] = lo[i]; r[8 + i] = hi[i]; }
  return r;
}

// B fragment from fragment-swizzled weights: tile (kb, obtile) stored as
// 32 lanes x 16 halfs contiguous (1 KB per tile, 32 B per lane).
__device__ inline v16h load_bw_frag(const _Float16* wsw, int nob, int kb,
                                    int obtile, int lane) {
  const _Float16* p = wsw + ((size_t)(kb * nob + obtile) << 9) + (lane << 4);
  v8h lo = *(const v8h*)p;
  v8h hi = *(const v8h*)(p + 8);
  v16h r;
#pragma unroll
  for (int i = 0; i < 8; ++i) { r[i] = lo[i]; r[8 + i] = hi[i]; }
  return r;
}

__device__ inline v8f wmma_f16(v16h a, v16h b, v8f c) {
  return __builtin_amdgcn_wmma_f32_16x16x32_f16(false, a, false, b, (short)0, c,
                                                false, false);
}

// ---------------------------------------------------------------------------
// Prep: fp32 row-major W[O][K] -> f16 fragment-swizzled layout.
// Tile (kb in K/32, ob in O/16) at base (kb*nob+ob)*512 halfs; within tile,
// lane l gets halfs i=0..15: element W[ob*16 + (l&15)][kb*32 + (l>>4)*16 + i].
__global__ void pack_w_kernel(const float* __restrict__ w, _Float16* __restrict__ wsw,
                              int O, int K) {
  int idx = blockIdx.x * blockDim.x + threadIdx.x;
  if (idx >= O * K) return;
  int nob  = O >> 4;
  int tile = idx >> 9;
  int rem  = idx & 511;
  int l    = rem >> 4;
  int i    = rem & 15;
  int kb   = tile / nob;
  int ob   = tile % nob;
  int o    = ob * 16 + (l & 15);
  int k    = kb * 32 + (l >> 4) * 16 + i;
  wsw[idx] = (_Float16)w[(size_t)o * K + k];
}

// Prep: qkv_bias = concat(q_bias, zeros, v_bias)
__global__ void qkvb_kernel(const float* __restrict__ qb, const float* __restrict__ vb,
                            float* __restrict__ out) {
  int i = blockIdx.x * blockDim.x + threadIdx.x;
  if (i >= 3 * DIM) return;
  out[i] = (i < DIM) ? qb[i] : (i >= 2 * DIM ? vb[i - 2 * DIM] : 0.0f);
}

// Prep: CPB relative-position bias -> bias[H][64][64] = 16*sigmoid(MLP(coords))
__global__ void cpb_kernel(const float* __restrict__ w1, const float* __restrict__ b1,
                           const float* __restrict__ w2, float* __restrict__ biasout) {
  __shared__ float tab[2 * WSZ - 1][HEADS];
  int t = threadIdx.x;
  if (t < 2 * WSZ - 1) {
    float c  = (float)(t - (WSZ - 1)) * (8.0f / (float)(WSZ - 1));
    float cs = copysignf(log2f(fabsf(c) + 1.0f) * (1.0f / 3.0f), c);
    float sum[HEADS];
#pragma unroll
    for (int h = 0; h < HEADS; ++h) sum[h] = 0.0f;
    for (int j = 0; j < 512; ++j) {
      float a = fmaxf(cs * w1[j] + b1[j], 0.0f);
#pragma unroll
      for (int h = 0; h < HEADS; ++h) sum[h] += a * w2[h * 512 + j];
    }
#pragma unroll
    for (int h = 0; h < HEADS; ++h) tab[t][h] = sum[h];
  }
  __syncthreads();
  for (int i = threadIdx.x; i < HEADS * WSZ * WSZ; i += blockDim.x) {
    int h = i >> 12, rem = i & 4095, row = rem >> 6, col = rem & 63;
    float v = tab[row - col + (WSZ - 1)][h];
    biasout[i] = 16.0f / (1.0f + __expf(-v));
  }
}

// ---------------------------------------------------------------------------
// Main: one block per window; 8 wave32 (wave = head for attention phases).
__global__ void __launch_bounds__(256, 1)
wattn_main(const float* __restrict__ x, const _Float16* __restrict__ wq_sw,
           const _Float16* __restrict__ wp_sw, const float* __restrict__ bias8,
           const float* __restrict__ logit_scale, const float* __restrict__ qkvb,
           const float* __restrict__ proj_b, float* __restrict__ out) {
  extern __shared__ _Float16 smem[];
  _Float16* xh   = smem;                 // 64 x XS : x staging, later attn output
  _Float16* qkvh = xh + WSZ * XS;        // 64 x QS : q | k
  _Float16* vT   = qkvh + WSZ * QS;      // 256 x VS : v transposed [dglobal][tok]
  _Float16* Pb   = vT + DIM * VS;        // 8 x 64 x PS : per-head probs

  const int bidx = blockIdx.x;
  const int tid  = threadIdx.x;
  const int lane = tid & 31;
  const int w    = tid >> 5;

  // ---- Phase 1: stage x as f16 (float4 loads, 8B LDS stores) ----
  const float* xb = x + (size_t)bidx * WSZ * DIM;
  for (int i4 = tid; i4 < WSZ * DIM / 4; i4 += 256) {
    v4f f = *(const v4f*)(xb + i4 * 4);
    int row = i4 >> 6, col = (i4 & 63) * 4;
    v4h hh;
#pragma unroll
    for (int j = 0; j < 4; ++j) hh[j] = (_Float16)f[j];
    *(v4h*)(xh + row * XS + col) = hh;
  }
  __syncthreads();

  // ---- Phase 2: QKV GEMM, M=64 N=768 K=256 ----
  for (int nt = 0; nt < 6; ++nt) {
    int obt = w * 6 + nt;  // wave-uniform output tile
    for (int mt = 0; mt < 4; ++mt) {
      v8f acc = {};
#pragma unroll
      for (int kb = 0; kb < 8; ++kb) {
        v16h a  = load_a_frag(xh, XS, mt * 16, kb * 32, lane);
        v16h bf = load_bw_frag(wq_sw, 48, kb, obt, lane);
        acc = wmma_f16(a, bf, acc);
      }
      int col   = obt * 16 + (lane & 15);
      float bv  = qkvb[col];
      int rbase = mt * 16 + (lane >> 4) * 8;
      if (obt < 32) {  // uniform branch: q|k tiles
#pragma unroll
        for (int r = 0; r < 8; ++r)
          qkvh[(rbase + r) * QS + col] = (_Float16)(acc[r] + bv);
      } else {         // v tiles: transposed store vT[dglobal][token]
        int d = col - 512;
#pragma unroll
        for (int r = 0; r < 8; ++r)
          vT[d * VS + rbase + r] = (_Float16)(acc[r] + bv);
      }
    }
  }
  __syncthreads();

  // ---- Phase 3: normalize q/k rows for this wave's head; fold logit scale ----
  const int h = w;
  _Float16* qh = qkvh + h * HD;
  _Float16* kh = qkvh + DIM + h * HD;
  float s = __expf(fminf(logit_scale[h], 4.6051701859880914f));  // ln(100)
  for (int rr = lane; rr < WSZ; rr += 32) {
    float sq = 0.0f, sk = 0.0f;
#pragma unroll
    for (int c = 0; c < HD; ++c) {
      float qv = (float)qh[rr * QS + c]; sq += qv * qv;
      float kv = (float)kh[rr * QS + c]; sk += kv * kv;
    }
    float iq = rsqrtf(sq) * s;
    float ik = rsqrtf(sk);
#pragma unroll
    for (int c = 0; c < HD; ++c) {
      qh[rr * QS + c] = (_Float16)((float)qh[rr * QS + c] * iq);
      kh[rr * QS + c] = (_Float16)((float)kh[rr * QS + c] * ik);
    }
  }
  // wave-private data from here on: no barrier needed

  // ---- Phase 4: logits = qn @ kn^T (64x64, K=32), bias, softmax ----
  v8f acc[4][4];
#pragma unroll
  for (int mt = 0; mt < 4; ++mt) {
    v16h a = load_a_frag(qh, QS, mt * 16, 0, lane);
#pragma unroll
    for (int nt = 0; nt < 4; ++nt) {
      v8f z = {};
      v16h bf = load_bt_frag(kh, QS, nt * 16, 0, lane);
      acc[mt][nt] = wmma_f16(a, bf, z);
    }
  }
  const float* bh = bias8 + (size_t)h * WSZ * WSZ;
#pragma unroll
  for (int mt = 0; mt < 4; ++mt)
#pragma unroll
    for (int nt = 0; nt < 4; ++nt)
#pragma unroll
      for (int r = 0; r < 8; ++r) {
        int row = mt * 16 + (lane >> 4) * 8 + r;
        int col = nt * 16 + (lane & 15);
        acc[mt][nt][r] += bh[row * WSZ + col];
      }
  _Float16* Ph = Pb + (size_t)h * WSZ * PS;
#pragma unroll
  for (int mt = 0; mt < 4; ++mt) {
#pragma unroll
    for (int r = 0; r < 8; ++r) {
      float mx = fmaxf(fmaxf(acc[mt][0][r], acc[mt][1][r]),
                       fmaxf(acc[mt][2][r], acc[mt][3][r]));
      mx = fmaxf(mx, __shfl_xor(mx, 1, 32));
      mx = fmaxf(mx, __shfl_xor(mx, 2, 32));
      mx = fmaxf(mx, __shfl_xor(mx, 4, 32));
      mx = fmaxf(mx, __shfl_xor(mx, 8, 32));
      float sum = 0.0f;
#pragma unroll
      for (int nt = 0; nt < 4; ++nt) {
        float e = __expf(acc[mt][nt][r] - mx);
        acc[mt][nt][r] = e;
        sum += e;
      }
      sum += __shfl_xor(sum, 1, 32);
      sum += __shfl_xor(sum, 2, 32);
      sum += __shfl_xor(sum, 4, 32);
      sum += __shfl_xor(sum, 8, 32);
      float inv = 1.0f / sum;
      int row = mt * 16 + (lane >> 4) * 8 + r;
#pragma unroll
      for (int nt = 0; nt < 4; ++nt)
        Ph[row * PS + nt * 16 + (lane & 15)] = (_Float16)(acc[mt][nt][r] * inv);
    }
  }

  // ---- Phase 5: out_h = P @ v  (M=64, N=32, K=64) ----
  _Float16* vTh = vT + (size_t)h * HD * VS;
#pragma unroll
  for (int mt = 0; mt < 4; ++mt)
#pragma unroll
    for (int nt = 0; nt < 2; ++nt) {
      v8f c = {};
#pragma unroll
      for (int kt = 0; kt < 2; ++kt) {
        v16h a  = load_a_frag(Ph, PS, mt * 16, kt * 32, lane);
        v16h bf = load_bt_frag(vTh, VS, nt * 16, kt * 32, lane);
        c = wmma_f16(a, bf, c);
      }
#pragma unroll
      for (int r = 0; r < 8; ++r) {
        int row = mt * 16 + (lane >> 4) * 8 + r;
        int col = h * HD + nt * 16 + (lane & 15);
        xh[row * XS + col] = (_Float16)c[r];  // reuse x staging as attn output
      }
    }
  __syncthreads();

  // ---- Phase 6: proj GEMM, M=64 N=256 K=256, fp32 out to HBM ----
  float* outb = out + (size_t)bidx * WSZ * DIM;
  for (int nt2 = 0; nt2 < 2; ++nt2) {
    int obt = w * 2 + nt2;
    for (int mt = 0; mt < 4; ++mt) {
      v8f c = {};
#pragma unroll
      for (int kb = 0; kb < 8; ++kb) {
        v16h a  = load_a_frag(xh, XS, mt * 16, kb * 32, lane);
        v16h bf = load_bw_frag(wp_sw, 16, kb, obt, lane);
        c = wmma_f16(a, bf, c);
      }
      int col  = obt * 16 + (lane & 15);
      float pb = proj_b[col];
#pragma unroll
      for (int r = 0; r < 8; ++r) {
        int row = mt * 16 + (lane >> 4) * 8 + r;
        outb[row * DIM + col] = c[r] + pb;
      }
    }
  }
}

// ---------------------------------------------------------------------------
extern "C" void kernel_launch(void* const* d_in, const int* in_sizes, int n_in,
                              void* d_out, int out_size, void* d_ws, size_t ws_size,
                              hipStream_t stream) {
  (void)in_sizes; (void)n_in; (void)out_size; (void)ws_size;
  const float* x           = (const float*)d_in[0];
  const float* qkv_w       = (const float*)d_in[1];
  const float* q_bias      = (const float*)d_in[2];
  const float* v_bias      = (const float*)d_in[3];
  const float* logit_scale = (const float*)d_in[4];
  const float* cpb_w1      = (const float*)d_in[5];
  const float* cpb_b1      = (const float*)d_in[6];
  const float* cpb_w2      = (const float*)d_in[7];
  const float* proj_w      = (const float*)d_in[8];
  const float* proj_b      = (const float*)d_in[9];
  float* out = (float*)d_out;

  char* ws = (char*)d_ws;
  _Float16* wq_sw = (_Float16*)ws;                               // 768*256*2 = 384 KB
  _Float16* wp_sw = (_Float16*)(ws + 393216);                    // 256*256*2 = 128 KB
  float*    biasb = (float*)(ws + 393216 + 131072);              // 8*64*64*4 = 128 KB
  float*    qkvb  = (float*)(ws + 393216 + 131072 + 131072);     // 768*4     =   3 KB

  pack_w_kernel<<<(768 * 256 + 255) / 256, 256, 0, stream>>>(qkv_w, wq_sw, 768, 256);
  pack_w_kernel<<<(256 * 256 + 255) / 256, 256, 0, stream>>>(proj_w, wp_sw, 256, 256);
  qkvb_kernel<<<3, 256, 0, stream>>>(q_bias, v_bias, qkvb);
  cpb_kernel<<<1, 128, 0, stream>>>(cpb_w1, cpb_b1, cpb_w2, biasb);

  size_t lds_bytes = (size_t)(WSZ * XS + WSZ * QS + DIM * VS + HEADS * WSZ * PS) *
                     sizeof(_Float16);  // ~206 KB, one WG per WGP
  wattn_main<<<4096, 256, lds_bytes, stream>>>(x, wq_sw, wp_sw, biasb, logit_scale,
                                               qkvb, proj_b, out);
}
